// Beta_score2_18872086298924
// MI455X (gfx1250) — compile-verified, matrix operand order checked by reading, not run on hardware
//
#include <hip/hip_runtime.h>
#include <hip/hip_bf16.h>

// ---------------------------------------------------------------------------
// Beta_score2: attention over 6 padded node-feature vectors.
//   scores[b,n] = h . tanh(W @ node_n[b] + bias)   (bf16 WMMA GEMM, K=D_n)
//   beta = softmax(scores, axis=n)
//   out[b, :] = sum_n beta[b,n] * padded_node_n[b, :]
// Roofline: ~192 MB min HBM traffic -> ~8us @ 23.3 TB/s; 34 GFLOP handled by
// v_wmma_f32_16x16x32_bf16 (only ~2.1M wmma issues device-wide). Node data
// (128 MB) fits in 192 MB L2, so the combine pass re-reads from L2.
// ---------------------------------------------------------------------------

typedef __attribute__((ext_vector_type(16))) __bf16 v16bf;
typedef __attribute__((ext_vector_type(8)))  float  v8f;

#define BATCH 8192
#define LDIM  2048
#define ODIM  512
#define NTILES 32          // ODIM/16
#define KSTEPS_W 64        // LDIM/32

__device__ __forceinline__ unsigned f2bf(float f) {
    unsigned u = __builtin_bit_cast(unsigned, f);
    unsigned r = u + 0x7FFFu + ((u >> 16) & 1u);   // round-to-nearest-even
    return r >> 16;
}
__device__ __forceinline__ unsigned pack2(float a, float b) {
    return f2bf(a) | (f2bf(b) << 16);
}

// ---------------------------------------------------------------------------
// Kernel A: convert W (512x2048 fp32, row-major) into bf16 blocks pre-swizzled
// for the WMMA B operand. Block (t,s) covers cols [16t,16t+16), K [32s,32s+32).
// Lane l (l<16: N=l, k in {0..7,16..23}; l>=16: N=l-16, k+8) holds 16 bf16 =
// 32 bytes contiguous -> GEMM loads it with two global_load_b128.
// ---------------------------------------------------------------------------
__global__ __launch_bounds__(32) void wswz_kernel(const float* __restrict__ W,
                                                  unsigned* __restrict__ wswz) {
    int blk = blockIdx.x;            // t*64 + s
    int t = blk >> 6, s = blk & 63;
    int l = threadIdx.x;             // 0..31
    int o = t * 16 + (l & 15);       // W row (output channel)
    const float* src = W + (size_t)o * LDIM + s * 32 + ((l >> 4) * 8);
    float4 f0 = *(const float4*)(src + 0);
    float4 f1 = *(const float4*)(src + 4);
    float4 f2 = *(const float4*)(src + 16);
    float4 f3 = *(const float4*)(src + 20);
    unsigned* dst = wswz + ((size_t)blk * 32 + l) * 8;
    uint4 p0 = make_uint4(pack2(f0.x, f0.y), pack2(f0.z, f0.w),
                          pack2(f1.x, f1.y), pack2(f1.z, f1.w));
    uint4 p1 = make_uint4(pack2(f2.x, f2.y), pack2(f2.z, f2.w),
                          pack2(f3.x, f3.y), pack2(f3.z, f3.w));
    *(uint4*)(dst + 0) = p0;
    *(uint4*)(dst + 4) = p1;
}

// ---------------------------------------------------------------------------
// Kernel B (templated on STEPS = K/32, compile-time trip count -> no remainder
// loop): scores_n[b] = h . tanh(W @ node + bias) for a 16-batch tile.
// grid = 512 batch-tiles, block = 128 (4 waves, full EXEC for WMMA).
// Stage the 16xK node tile into LDS in bf16 WMMA A-layout, then each wave
// owns 8 N-tiles and runs the K loop of v_wmma_f32_16x16x32_bf16.
// ---------------------------------------------------------------------------
template <int STEPS>
__global__ __launch_bounds__(128) void score_kernel(
    const float* __restrict__ src,
    const unsigned* __restrict__ wswz,
    const float* __restrict__ bias, const float* __restrict__ hn,
    float* __restrict__ scores_n) {

    constexpr int K = STEPS * 32;

    __shared__ __align__(32) unsigned ldsA[STEPS * 32 * 8];  // <= 32KB
    __shared__ float ssc[16];

    int bt  = blockIdx.x;          // batch tile (16 rows)
    int tid = threadIdx.x;
    if (tid < 16) ssc[tid] = 0.0f;

    // ---- stage node tile -> LDS bf16, WMMA A layout --------------------
    // item id = (s, l): lane l of K-step s; 16 bf16 (32B) per item.
#pragma unroll
    for (int id = tid; id < STEPS * 32; id += 128) {
        int s = id >> 5, l = id & 31;
        int m = l & 15;                                   // batch row in tile
        const float* p = src + (size_t)(bt * 16 + m) * K + s * 32 + (l >> 4) * 8;
        float4 f0 = *(const float4*)(p + 0);
        float4 f1 = *(const float4*)(p + 4);
        float4 f2 = *(const float4*)(p + 16);
        float4 f3 = *(const float4*)(p + 20);
        unsigned* dst = &ldsA[(s * 32 + l) * 8];
        uint4 p0 = make_uint4(pack2(f0.x, f0.y), pack2(f0.z, f0.w),
                              pack2(f1.x, f1.y), pack2(f1.z, f1.w));
        uint4 p1 = make_uint4(pack2(f2.x, f2.y), pack2(f2.z, f2.w),
                              pack2(f3.x, f3.y), pack2(f3.z, f3.w));
        *(uint4*)(dst + 0) = p0;
        *(uint4*)(dst + 4) = p1;
    }
    __syncthreads();

    // ---- WMMA GEMM + fused tanh/h-dot reduction ------------------------
    int w = tid >> 5;             // wave id 0..3 -> N-tiles [8w, 8w+8)
    int l = tid & 31;
    int half = l >> 4, lcol = l & 15;

    float part[8];
#pragma unroll
    for (int i = 0; i < 8; ++i) part[i] = 0.0f;

    for (int t = w * 8; t < w * 8 + 8; ++t) {
        int col = t * 16 + lcol;                // output channel for this lane
        float bc = bias[col];
        float hc = hn[col];
        v8f acc = {};
        const unsigned* wb = wswz + ((size_t)(t * 64) * 32 + l) * 8;
#pragma unroll 4
        for (int s = 0; s < STEPS; ++s) {
            v16bf a  = *(const v16bf*)&ldsA[(s * 32 + l) * 8];
            v16bf bm = *(const v16bf*)(wb + (size_t)s * 32 * 8);
            acc = __builtin_amdgcn_wmma_f32_16x16x32_bf16(
                false, a, false, bm, (short)0, acc, false, false);
        }
#pragma unroll
        for (int i = 0; i < 8; ++i)
            part[i] += tanhf(acc[i] + bc) * hc;
    }

    // reduce over the 16 lanes that share the same row-half (xor stays in half)
#pragma unroll
    for (int i = 0; i < 8; ++i) {
        float v = part[i];
        v += __shfl_xor(v, 1, 32);
        v += __shfl_xor(v, 2, 32);
        v += __shfl_xor(v, 4, 32);
        v += __shfl_xor(v, 8, 32);
        part[i] = v;
    }
    if (lcol == 0) {
#pragma unroll
        for (int i = 0; i < 8; ++i)
            atomicAdd(&ssc[half * 8 + i], part[i]);   // rows 0..7 / 8..15
    }
    __syncthreads();

    if (tid < 16)
        scores_n[bt * 16 + tid] = ssc[tid];
}

// ---------------------------------------------------------------------------
// Kernel C: per batch row -- softmax over the 6 scores, then float4-vectorized
// weighted sum of the (virtually padded) node features. Node data is L2-hot.
// ---------------------------------------------------------------------------
__global__ __launch_bounds__(256) void combine_kernel(
    const float* __restrict__ n0, const float* __restrict__ n1,
    const float* __restrict__ n2, const float* __restrict__ n3,
    const float* __restrict__ n4, const float* __restrict__ n5,
    const float* __restrict__ scores, float* __restrict__ out) {

    int b = blockIdx.x;
    int tid = threadIdx.x;

    float s[6];
#pragma unroll
    for (int n = 0; n < 6; ++n) s[n] = scores[(size_t)n * BATCH + b];
    float m = s[0];
#pragma unroll
    for (int n = 1; n < 6; ++n) m = fmaxf(m, s[n]);
    float e[6], sum = 0.0f;
#pragma unroll
    for (int n = 0; n < 6; ++n) { e[n] = __expf(s[n] - m); sum += e[n]; }
    float inv = 1.0f / sum;
    float b0 = e[0] * inv, b1 = e[1] * inv, b2 = e[2] * inv,
          b3 = e[3] * inv, b4 = e[4] * inv, b5 = e[5] * inv;

    // 2048 cols = 512 float4 chunks; 256 threads * 2 chunks.
    // chunk tid      -> cols [0,1024): full / two-term region
    // chunk tid+256  -> cols [1024,2048): exact zeros
    {
        int col = tid * 4;
        float4 r;
        if (col < 512) {
            float4 a0 = *(const float4*)(n0 + (size_t)b * 1024 + col);
            float4 a1 = *(const float4*)(n1 + (size_t)b * 512 + col);
            float4 a2 = *(const float4*)(n2 + (size_t)b * 512 + col);
            float4 a3 = *(const float4*)(n3 + (size_t)b * 512 + col);
            float4 a4 = *(const float4*)(n4 + (size_t)b * 1024 + col);
            float4 a5 = *(const float4*)(n5 + (size_t)b * 512 + col);
            r.x = b0 * a0.x + b1 * a1.x + b2 * a2.x + b3 * a3.x + b4 * a4.x + b5 * a5.x;
            r.y = b0 * a0.y + b1 * a1.y + b2 * a2.y + b3 * a3.y + b4 * a4.y + b5 * a5.y;
            r.z = b0 * a0.z + b1 * a1.z + b2 * a2.z + b3 * a3.z + b4 * a4.z + b5 * a5.z;
            r.w = b0 * a0.w + b1 * a1.w + b2 * a2.w + b3 * a3.w + b4 * a4.w + b5 * a5.w;
        } else {   // cols [512,1024): only ls (D=1024) and ds (D=1024)
            float4 a0 = *(const float4*)(n0 + (size_t)b * 1024 + col);
            float4 a4 = *(const float4*)(n4 + (size_t)b * 1024 + col);
            r.x = b0 * a0.x + b4 * a4.x;
            r.y = b0 * a0.y + b4 * a4.y;
            r.z = b0 * a0.z + b4 * a4.z;
            r.w = b0 * a0.w + b4 * a4.w;
        }
        *(float4*)(out + (size_t)b * LDIM + col) = r;
    }
    {
        int col = (tid + 256) * 4;                 // >= 1024: padded zeros
        float4 z = make_float4(0.f, 0.f, 0.f, 0.f);
        *(float4*)(out + (size_t)b * LDIM + col) = z;
    }
}

// ---------------------------------------------------------------------------
extern "C" void kernel_launch(void* const* d_in, const int* in_sizes, int n_in,
                              void* d_out, int out_size, void* d_ws, size_t ws_size,
                              hipStream_t stream) {
    const float* ls  = (const float*)d_in[0];   // [8192,1,1024]
    const float* A   = (const float*)d_in[1];   // [8192,1,512]
    const float* lm  = (const float*)d_in[2];   // [8192,1,512]
    const float* AT  = (const float*)d_in[3];   // [8192,1,512]
    const float* ds  = (const float*)d_in[4];   // [8192,1,1024]
    const float* dm  = (const float*)d_in[5];   // [8192,1,512]
    const float* W   = (const float*)d_in[6];   // [512,2048]
    const float* bia = (const float*)d_in[7];   // [512]
    const float* hn  = (const float*)d_in[8];   // [512,1]
    float* out = (float*)d_out;

    // workspace: [0, 2MB) swizzled bf16 W; [2MB, +192KB) scores (6*8192 f32)
    unsigned* wswz = (unsigned*)d_ws;
    float* scores = (float*)((char*)d_ws + (size_t)2 * 1024 * 1024);

    wswz_kernel<<<NTILES * KSTEPS_W, 32, 0, stream>>>(W, wswz);

    // one launch per node type; STEPS = D_n/32 is compile-time
    score_kernel<32><<<BATCH / 16, 128, 0, stream>>>(ls, wswz, bia, hn, scores + 0 * BATCH);
    score_kernel<16><<<BATCH / 16, 128, 0, stream>>>(A,  wswz, bia, hn, scores + 1 * BATCH);
    score_kernel<16><<<BATCH / 16, 128, 0, stream>>>(lm, wswz, bia, hn, scores + 2 * BATCH);
    score_kernel<16><<<BATCH / 16, 128, 0, stream>>>(AT, wswz, bia, hn, scores + 3 * BATCH);
    score_kernel<32><<<BATCH / 16, 128, 0, stream>>>(ds, wswz, bia, hn, scores + 4 * BATCH);
    score_kernel<16><<<BATCH / 16, 128, 0, stream>>>(dm, wswz, bia, hn, scores + 5 * BATCH);

    combine_kernel<<<BATCH, 256, 0, stream>>>(ls, A, lm, AT, ds, dm,
                                              scores, out);
}
